// WanAnimateAttentionBlock_23149873726015
// MI455X (gfx1250) — matrix operand order repaired and checked by reading, not compiled, not run on hardware
//
#include <hip/hip_runtime.h>
#include <hip/hip_bf16.h>

// ---------------------------------------------------------------------------
// WanAnimate attention block for MI455X (gfx1250, wave32, WMMA).
// All heavy math runs through v_wmma_f32_16x16x32_bf16 (fp32 accumulate).
// GEMM: double-buffered LDS + register prefetch + global_prefetch for weights.
// Attention: batched fragment loads so WMMAs overlap load latency.
// ---------------------------------------------------------------------------

#define B_ 1
#define L_ 3072
#define C_ 2048
#define NH_ 16
#define HD_ 128
#define FFN_ 8192
#define LIMG_ 257
#define LTXT_ 512
#define EPSF 1e-6f

typedef __bf16 bf16_t;
typedef __attribute__((ext_vector_type(16))) __bf16 v16bf;
typedef __attribute__((ext_vector_type(8)))  float  v8f;

// ---------------- small helpers ----------------
static __device__ __forceinline__ bf16_t f2bf(float f) {
    unsigned int u = __builtin_bit_cast(unsigned int, f);
    u += 0x7FFFu + ((u >> 16) & 1u);           // round-to-nearest-even
    unsigned short s = (unsigned short)(u >> 16);
    return __builtin_bit_cast(bf16_t, s);
}

struct alignas(16) FragBits { uint4 a, b; };

// load a v16bf WMMA fragment from two 16B chunks (global or LDS)
template <typename P>
static __device__ __forceinline__ v16bf load_frag(const P* p0, const P* p1) {
    FragBits fb;
    fb.a = *(const uint4*)p0;
    fb.b = *(const uint4*)p1;
    return __builtin_bit_cast(v16bf, fb);
}

static __device__ __forceinline__ v8f wmma_bf16(v16bf a, v16bf b, v8f c) {
    return __builtin_amdgcn_wmma_f32_16x16x32_bf16(false, a, false, b,
                                                   (short)0, c, false, false);
}

static __device__ __forceinline__ float rmax16(float v) {
    v = fmaxf(v, __shfl_xor(v, 1, 32));
    v = fmaxf(v, __shfl_xor(v, 2, 32));
    v = fmaxf(v, __shfl_xor(v, 4, 32));
    v = fmaxf(v, __shfl_xor(v, 8, 32));
    return v;
}
static __device__ __forceinline__ float rsum16(float v) {
    v += __shfl_xor(v, 1, 32);
    v += __shfl_xor(v, 2, 32);
    v += __shfl_xor(v, 4, 32);
    v += __shfl_xor(v, 8, 32);
    return v;
}

static __device__ __forceinline__ float gelu_tanh(float x) {
    float x3 = x * x * x;
    return 0.5f * x * (1.0f + tanhf(0.7978845608028654f * (x + 0.044715f * x3)));
}

// ---------------------------------------------------------------------------
// GEMM: D[M,N] = A_bf16[M,K] * W_f32[K,N] + bias ; optional GELU ; out f32/bf16
// Block: 256 threads (8 waves). Tile 128x128x32. Wave = 32x64 (2x4 WMMA tiles).
// Double-buffered LDS; next tile prefetched into registers during compute.
// ---------------------------------------------------------------------------
template <int GELU, int OUT_BF16>
__global__ __launch_bounds__(256) void gemm_kernel(
    const bf16_t* __restrict__ A, const float* __restrict__ Bw,
    const float* __restrict__ bias, void* __restrict__ Cout,
    int M, int N, int K)
{
    __shared__ alignas(16) bf16_t As[2][128][40];  // padded rows: conflict-free
    __shared__ alignas(16) bf16_t Bt[2][128][40];  // B^T tile: [n][k]

    const int t    = threadIdx.x;
    const int w    = t >> 5;
    const int lane = t & 31;
    const int half = lane >> 4;
    const int l15  = lane & 15;
    const int m0   = blockIdx.y * 128;
    const int n0   = blockIdx.x * 128;
    const int wm   = (w & 3) * 32;
    const int wn   = (w >> 2) * 64;

    v8f acc[2][4];
#pragma unroll
    for (int i = 0; i < 2; ++i)
#pragma unroll
        for (int j = 0; j < 4; ++j)
            acc[i][j] = (v8f){0.f,0.f,0.f,0.f,0.f,0.f,0.f,0.f};

    const int ar = t >> 1;            // A loader: row, 16 bf16 per thread
    const int ac = (t & 1) * 16;
    const int bk = t >> 3;            // B loader: k row, 16 f32 per thread
    const int bc = (t & 7) * 16;
    const bool arow_ok = (m0 + ar) < M;

    uint4  ra0, ra1;                  // register-staged A tile chunk
    float4 rb[4];                     // register-staged B tile chunk

    auto fetch = [&](int k0) {
        if (arow_ok) {
            const bf16_t* ag = A + (size_t)(m0 + ar) * K + k0 + ac;
            ra0 = *(const uint4*)ag;
            ra1 = *(const uint4*)(ag + 8);
        } else {
            ra0 = (uint4){0,0,0,0}; ra1 = (uint4){0,0,0,0};
        }
        const float* bg = Bw + (size_t)(k0 + bk) * N + n0 + bc;
#pragma unroll
        for (int q4 = 0; q4 < 4; ++q4)
            rb[q4] = *(const float4*)(bg + q4 * 4);
    };
    auto stage = [&](int buf) {
        *(uint4*)&As[buf][ar][ac]     = ra0;
        *(uint4*)&As[buf][ar][ac + 8] = ra1;
#pragma unroll
        for (int q4 = 0; q4 < 4; ++q4) {
            Bt[buf][bc + q4 * 4 + 0][bk] = f2bf(rb[q4].x);
            Bt[buf][bc + q4 * 4 + 1][bk] = f2bf(rb[q4].y);
            Bt[buf][bc + q4 * 4 + 2][bk] = f2bf(rb[q4].z);
            Bt[buf][bc + q4 * 4 + 3][bk] = f2bf(rb[q4].w);
        }
    };

    const int nk = K >> 5;
    fetch(0);
    stage(0);
    __syncthreads();

    for (int s = 0; s < nk; ++s) {
        const int buf = s & 1;
        // prefetch next tile into registers; warm L2 two tiles ahead
        if (s + 1 < nk) fetch((s + 1) * 32);
        if (s + 2 < nk) {
            __builtin_prefetch(Bw + (size_t)((s + 2) * 32 + bk) * N + n0 + bc, 0, 1);
            if (arow_ok)
                __builtin_prefetch(A + (size_t)(m0 + ar) * K + (s + 2) * 32 + ac, 0, 1);
        }

        v16bf af[2], bfm[4];
#pragma unroll
        for (int i = 0; i < 2; ++i) {
            int r = wm + i * 16 + l15;
            af[i] = load_frag(&As[buf][r][half * 8], &As[buf][r][16 + half * 8]);
        }
#pragma unroll
        for (int j = 0; j < 4; ++j) {
            int n = wn + j * 16 + l15;
            bfm[j] = load_frag(&Bt[buf][n][half * 8], &Bt[buf][n][16 + half * 8]);
        }
#pragma unroll
        for (int i = 0; i < 2; ++i)
#pragma unroll
            for (int j = 0; j < 4; ++j)
                acc[i][j] = wmma_bf16(af[i], bfm[j], acc[i][j]);

        if (s + 1 < nk) stage((s + 1) & 1);   // other buffer: free since s-1
        __syncthreads();
    }

    // ---- epilogue: bias (+gelu), store f32 or bf16 ----
#pragma unroll
    for (int i = 0; i < 2; ++i)
#pragma unroll
        for (int j = 0; j < 4; ++j) {
            int col  = n0 + wn + j * 16 + l15;
            float bs = bias[col];
#pragma unroll
            for (int r = 0; r < 8; ++r) {
                int row = m0 + wm + i * 16 + r + half * 8;
                if (row < M) {
                    float v = acc[i][j][r] + bs;
                    if (GELU) v = gelu_tanh(v);
                    if (OUT_BF16)
                        ((bf16_t*)Cout)[(size_t)row * N + col] = f2bf(v);
                    else
                        ((float*)Cout)[(size_t)row * N + col] = v;
                }
            }
        }
}

// ---------------------------------------------------------------------------
// Flash attention: Q[Lq,NH*HD] x K[Lk,..] -> softmax -> x V[Lk,..]  (bf16 in,
// f32 out). Block = 256 thr (8 waves), 128 q rows / block, 16 q rows / wave.
// kv chunks of 32; S and P*V both via WMMA bf16. K fragments are batch-loaded
// so the 8 S-WMMAs overlap global load latency instead of serializing on it.
// ---------------------------------------------------------------------------
__global__ __launch_bounds__(256) void attn_kernel(
    const bf16_t* __restrict__ Q, const bf16_t* __restrict__ K,
    const bf16_t* __restrict__ V, float* __restrict__ O,
    int Lk, float scale)
{
    __shared__ alignas(16) bf16_t Vt[128][40];       // V^T chunk: [hd][kv]
    __shared__ alignas(16) bf16_t Pl[8][16][40];     // per-wave P staging

    const int t    = threadIdx.x;
    const int w    = t >> 5;
    const int lane = t & 31;
    const int half = lane >> 4;
    const int l15  = lane & 15;
    const int head = blockIdx.y;
    const int q0   = blockIdx.x * 128 + w * 16;

    // Q fragments for this wave's 16 rows (HD=128 -> 4 K-chunks of 32)
    v16bf qf[4];
#pragma unroll
    for (int c = 0; c < 4; ++c) {
        const bf16_t* qp = Q + (size_t)(q0 + l15) * C_ + head * HD_ + c * 32;
        qf[c] = load_frag(qp + half * 8, qp + 16 + half * 8);
    }

    v8f acc[8];
#pragma unroll
    for (int tt = 0; tt < 8; ++tt)
        acc[tt] = (v8f){0.f,0.f,0.f,0.f,0.f,0.f,0.f,0.f};
    float mrow[8], lrow[8];
#pragma unroll
    for (int r = 0; r < 8; ++r) { mrow[r] = -1e30f; lrow[r] = 0.f; }

    const int kvr = t >> 3;            // V^T loader: kv row
    const int hg  = (t & 7) * 16;      // 16 hd values
    const int nch = (Lk + 31) / 32;

    for (int ch = 0; ch < nch; ++ch) {
        const int kb = ch * 32;
        __syncthreads();               // previous chunk's Vt fully consumed
        // ---- stage V^T in LDS (zero-pad past Lk) ----
        {
            bf16_t tmp[16];
            int krow = kb + kvr;
            if (krow < Lk) {
                const bf16_t* vp = V + (size_t)krow * C_ + head * HD_ + hg;
                *(uint4*)&tmp[0] = *(const uint4*)vp;
                *(uint4*)&tmp[8] = *(const uint4*)(vp + 8);
            } else {
                bf16_t z = f2bf(0.f);
#pragma unroll
                for (int jj = 0; jj < 16; ++jj) tmp[jj] = z;
            }
#pragma unroll
            for (int jj = 0; jj < 16; ++jj) Vt[hg + jj][kvr] = tmp[jj];
        }
        __syncthreads();

        // ---- S = Q K^T for two 16-wide kv tiles: batch all loads first ----
        v16bf kf[2][4];
#pragma unroll
        for (int kt = 0; kt < 2; ++kt) {
            const bf16_t* kp = K + (size_t)(kb + kt * 16 + l15) * C_ +
                               head * HD_ + half * 8;
#pragma unroll
            for (int c = 0; c < 4; ++c)
                kf[kt][c] = load_frag(kp + c * 32, kp + c * 32 + 16);
        }
        v8f s[2];
        s[0] = (v8f){0.f,0.f,0.f,0.f,0.f,0.f,0.f,0.f};
        s[1] = (v8f){0.f,0.f,0.f,0.f,0.f,0.f,0.f,0.f};
#pragma unroll
        for (int kt = 0; kt < 2; ++kt)
#pragma unroll
            for (int c = 0; c < 4; ++c)
                s[kt] = wmma_bf16(qf[c], kf[kt][c], s[kt]);

        // scale + mask (kv >= Lk)
#pragma unroll
        for (int kt = 0; kt < 2; ++kt) {
            bool oob = (kb + kt * 16 + l15) >= Lk;
#pragma unroll
            for (int r = 0; r < 8; ++r) {
                float v = s[kt][r] * scale;
                s[kt][r] = oob ? -1e30f : v;
            }
        }
        // ---- online softmax, stage P (bf16) in LDS, rescale acc ----
#pragma unroll
        for (int r = 0; r < 8; ++r) {
            float a  = fmaxf(rmax16(s[0][r]), rmax16(s[1][r]));
            float mn = fmaxf(mrow[r], a);
            float alpha = __expf(mrow[r] - mn);
            mrow[r] = mn;
            float p0 = __expf(s[0][r] - mn);
            float p1 = __expf(s[1][r] - mn);
            lrow[r] = lrow[r] * alpha + rsum16(p0 + p1);
            int prow = r + half * 8;
            Pl[w][prow][l15]      = f2bf(p0);
            Pl[w][prow][16 + l15] = f2bf(p1);
#pragma unroll
            for (int tt = 0; tt < 8; ++tt) acc[tt][r] *= alpha;
        }
        __syncthreads();               // make P + Vt visible for the MMAs

        // ---- acc += P * V ----
        const bf16_t* pr = &Pl[w][l15][0];
        v16bf pf = load_frag(pr + half * 8, pr + 16 + half * 8);
#pragma unroll
        for (int tt = 0; tt < 8; ++tt) {
            int n = tt * 16 + l15;
            v16bf vf = load_frag(&Vt[n][half * 8], &Vt[n][16 + half * 8]);
            acc[tt] = wmma_bf16(pf, vf, acc[tt]);
        }
    }

    // ---- normalize + store f32 ----
#pragma unroll
    for (int tt = 0; tt < 8; ++tt)
#pragma unroll
        for (int r = 0; r < 8; ++r) {
            int row = q0 + r + half * 8;
            int col = head * HD_ + tt * 16 + l15;
            O[(size_t)row * C_ + col] = acc[tt][r] / lrow[r];
        }
}

// ---------------------------------------------------------------------------
// Elementwise / normalization kernels
// ---------------------------------------------------------------------------
__global__ __launch_bounds__(256) void es_kernel(
    const float* __restrict__ mod, const float* __restrict__ e,
    float* __restrict__ es, int n)
{
    int i = blockIdx.x * 256 + threadIdx.x;
    if (i < n) es[i] = mod[i] + e[i];
}

// LN over C_=2048, optional affine (w,b) and modulation (*(1+sc)+sh), bf16 out
__global__ __launch_bounds__(256) void ln_mod_kernel(
    const float* __restrict__ X, const float* __restrict__ w,
    const float* __restrict__ b, const float* __restrict__ sc,
    const float* __restrict__ sh, bf16_t* __restrict__ out)
{
    __shared__ float s1[256], s2[256];
    const int row = blockIdx.x, t = threadIdx.x;
    const float* xr = X + (size_t)row * C_;
    float xv[8], su = 0.f, sq = 0.f;
#pragma unroll
    for (int i = 0; i < 8; ++i) {
        xv[i] = xr[t + i * 256];
        su += xv[i];
        sq += xv[i] * xv[i];
    }
    s1[t] = su; s2[t] = sq;
    __syncthreads();
    for (int off = 128; off > 0; off >>= 1) {
        if (t < off) { s1[t] += s1[t + off]; s2[t] += s2[t + off]; }
        __syncthreads();
    }
    float mean = s1[0] * (1.0f / C_);
    float var  = s2[0] * (1.0f / C_) - mean * mean;
    float rstd = rsqrtf(var + EPSF);
#pragma unroll
    for (int i = 0; i < 8; ++i) {
        int c = t + i * 256;
        float y = (xv[i] - mean) * rstd;
        if (w)  y = y * w[c] + b[c];
        if (sc) y = y * (1.0f + sc[c]) + sh[c];
        out[(size_t)row * C_ + c] = f2bf(y);
    }
}

// RMS norm over C_=2048, in-place f32
__global__ __launch_bounds__(256) void rms_kernel(
    float* __restrict__ X, const float* __restrict__ w)
{
    __shared__ float s2[256];
    const int row = blockIdx.x, t = threadIdx.x;
    float* xr = X + (size_t)row * C_;
    float xv[8], sq = 0.f;
#pragma unroll
    for (int i = 0; i < 8; ++i) {
        xv[i] = xr[t + i * 256];
        sq += xv[i] * xv[i];
    }
    s2[t] = sq;
    __syncthreads();
    for (int off = 128; off > 0; off >>= 1) {
        if (t < off) s2[t] += s2[t + off];
        __syncthreads();
    }
    float r = rsqrtf(s2[0] * (1.0f / C_) + EPSF);
#pragma unroll
    for (int i = 0; i < 8; ++i) {
        int c = t + i * 256;
        xr[c] = xv[i] * r * w[c];
    }
}

// RoPE on f32 [L,NH,HD] -> bf16; freqs table [1024,64]
__global__ __launch_bounds__(256) void rope_kernel(
    const float* __restrict__ in, const float* __restrict__ freqs,
    bf16_t* __restrict__ out)
{
    int pid = blockIdx.x * 256 + threadIdx.x;
    if (pid >= L_ * NH_ * 64) return;
    int d    = pid & 63;
    int row  = pid >> 10;          // 16 heads * 64 pairs = 1024 per token
    int f    = row >> 10;          // 32*32 tokens per frame
    int hh   = (row >> 5) & 31;
    int ww   = row & 31;
    int pos  = d < 22 ? f : (d < 43 ? hh : ww);
    float ang = freqs[pos * 64 + d];
    float sn, cs;
    __sincosf(ang, &sn, &cs);
    size_t base = (size_t)pid * 2;  // == row*C_ + head*HD_ + d*2
    float x0 = in[base], x1 = in[base + 1];
    out[base]     = f2bf(x0 * cs - x1 * sn);
    out[base + 1] = f2bf(x0 * sn + x1 * cs);
}

// out_bf16 = bf16(a + (b ? b : 0))
__global__ __launch_bounds__(256) void addcvt_kernel(
    const float* __restrict__ a, const float* __restrict__ b,
    bf16_t* __restrict__ out, int n)
{
    int i = blockIdx.x * 256 + threadIdx.x;
    if (i < n) out[i] = f2bf(a[i] + (b ? b[i] : 0.f));
}

// out = x + y * (gate ? gate[c] : 1)
__global__ __launch_bounds__(256) void residual_kernel(
    const float* __restrict__ x, const float* __restrict__ y,
    const float* __restrict__ gate, float* __restrict__ out, int n)
{
    int i = blockIdx.x * 256 + threadIdx.x;
    if (i < n) {
        float g = gate ? gate[i & (C_ - 1)] : 1.0f;
        out[i] = x[i] + y[i] * g;
    }
}

// ---------------------------------------------------------------------------
// Host launch
// ---------------------------------------------------------------------------
extern "C" void kernel_launch(void* const* d_in, const int* in_sizes, int n_in,
                              void* d_out, int out_size, void* d_ws, size_t ws_size,
                              hipStream_t stream)
{
    (void)in_sizes; (void)n_in; (void)out_size; (void)ws_size;
    const float* x       = (const float*)d_in[0];
    const float* e       = (const float*)d_in[1];
    const float* freqs   = (const float*)d_in[4];
    const float* context = (const float*)d_in[5];
    const float* modw    = (const float*)d_in[7];
    const float* n3w     = (const float*)d_in[8];
    const float* n3b     = (const float*)d_in[9];
    const float* sa_qw = (const float*)d_in[10], *sa_qb = (const float*)d_in[11];
    const float* sa_kw = (const float*)d_in[12], *sa_kb = (const float*)d_in[13];
    const float* sa_vw = (const float*)d_in[14], *sa_vb = (const float*)d_in[15];
    const float* sa_ow = (const float*)d_in[16], *sa_ob = (const float*)d_in[17];
    const float* sa_nq = (const float*)d_in[18], *sa_nk = (const float*)d_in[19];
    const float* ca_qw = (const float*)d_in[20], *ca_qb = (const float*)d_in[21];
    const float* ca_kw = (const float*)d_in[22], *ca_kb = (const float*)d_in[23];
    const float* ca_vw = (const float*)d_in[24], *ca_vb = (const float*)d_in[25];
    const float* ca_ow = (const float*)d_in[26], *ca_ob = (const float*)d_in[27];
    const float* ca_kiw = (const float*)d_in[28], *ca_kib = (const float*)d_in[29];
    const float* ca_viw = (const float*)d_in[30], *ca_vib = (const float*)d_in[31];
    const float* ca_nq  = (const float*)d_in[32], *ca_nk = (const float*)d_in[33];
    const float* ca_nki = (const float*)d_in[34];
    const float* f_w1 = (const float*)d_in[35], *f_b1 = (const float*)d_in[36];
    const float* f_w2 = (const float*)d_in[37], *f_b2 = (const float*)d_in[38];

    // ---- carve workspace ----
    char*  base = (char*)d_ws;
    size_t off  = 0;
    auto alloc = [&](size_t bytes) -> char* {
        char* p = base + off;
        off += (bytes + 255) & ~(size_t)255;
        return p;
    };
    const size_t LC = (size_t)L_ * C_;
    float*  es   = (float*)alloc(6 * C_ * 4);
    float*  f0   = (float*)alloc(LC * 4);
    float*  f1   = (float*)alloc(LC * 4);
    float*  f2   = (float*)alloc(LC * 4);
    float*  f3   = (float*)alloc(LC * 4);
    bf16_t* b_h  = (bf16_t*)alloc(LC * 2);
    bf16_t* b_q  = (bf16_t*)alloc(LC * 2);
    bf16_t* b_k  = (bf16_t*)alloc(LC * 2);
    bf16_t* b_v  = (bf16_t*)alloc(LC * 2);
    bf16_t* b_ao = (bf16_t*)alloc(LC * 2);
    bf16_t* b_xn = (bf16_t*)alloc(LC * 2);
    bf16_t* b_ffn = (bf16_t*)alloc((size_t)L_ * FFN_ * 2);
    bf16_t* ctxt_bf = (bf16_t*)alloc((size_t)LTXT_ * C_ * 2);
    bf16_t* cimg_bf = (bf16_t*)alloc((size_t)288 * C_ * 2);
    bf16_t* ck_bf   = (bf16_t*)alloc((size_t)LTXT_ * C_ * 2);
    bf16_t* cv_bf   = (bf16_t*)alloc((size_t)LTXT_ * C_ * 2);
    bf16_t* cik_bf  = (bf16_t*)alloc((size_t)288 * C_ * 2);
    bf16_t* civ_bf  = (bf16_t*)alloc((size_t)288 * C_ * 2);

    const int n_lc = (int)LC;
    const dim3 blk(256);
    const dim3 g_ew((n_lc + 255) / 256);
    const float scale = 0.08838834764831845f;   // HD^-0.5
    auto gemm_grid = [](int M, int N) { return dim3(N / 128, (M + 127) / 128); };

    // es = modulation + e
    es_kernel<<<dim3((6 * C_ + 255) / 256), blk, 0, stream>>>(modw, e, es, 6 * C_);

    // ---- self attention ----
    ln_mod_kernel<<<dim3(L_), blk, 0, stream>>>(x, nullptr, nullptr,
                                                es + 1 * C_, es + 0 * C_, b_h);
    gemm_kernel<0,0><<<gemm_grid(L_, C_), blk, 0, stream>>>(b_h, sa_qw, sa_qb, f0, L_, C_, C_);
    gemm_kernel<0,0><<<gemm_grid(L_, C_), blk, 0, stream>>>(b_h, sa_kw, sa_kb, f1, L_, C_, C_);
    gemm_kernel<0,0><<<gemm_grid(L_, C_), blk, 0, stream>>>(b_h, sa_vw, sa_vb, f2, L_, C_, C_);
    rms_kernel<<<dim3(L_), blk, 0, stream>>>(f0, sa_nq);
    rms_kernel<<<dim3(L_), blk, 0, stream>>>(f1, sa_nk);
    {
        int np = L_ * NH_ * 64;
        rope_kernel<<<dim3((np + 255) / 256), blk, 0, stream>>>(f0, freqs, b_q);
        rope_kernel<<<dim3((np + 255) / 256), blk, 0, stream>>>(f1, freqs, b_k);
    }
    addcvt_kernel<<<g_ew, blk, 0, stream>>>(f2, nullptr, b_v, n_lc);
    attn_kernel<<<dim3(L_ / 128, NH_), blk, 0, stream>>>(b_q, b_k, b_v, f0, L_, scale);
    addcvt_kernel<<<g_ew, blk, 0, stream>>>(f0, nullptr, b_ao, n_lc);
    gemm_kernel<0,0><<<gemm_grid(L_, C_), blk, 0, stream>>>(b_ao, sa_ow, sa_ob, f1, L_, C_, C_);
    residual_kernel<<<g_ew, blk, 0, stream>>>(x, f1, es + 2 * C_, f3, n_lc);

    // ---- cross attention ----
    ln_mod_kernel<<<dim3(L_), blk, 0, stream>>>(f3, n3w, n3b, nullptr, nullptr, b_xn);
    {
        int nt = LTXT_ * C_, ni = LIMG_ * C_;
        addcvt_kernel<<<dim3((nt + 255) / 256), blk, 0, stream>>>(
            context + (size_t)LIMG_ * C_, nullptr, ctxt_bf, nt);
        addcvt_kernel<<<dim3((ni + 255) / 256), blk, 0, stream>>>(
            context, nullptr, cimg_bf, ni);
    }
    gemm_kernel<0,0><<<gemm_grid(L_, C_), blk, 0, stream>>>(b_xn, ca_qw, ca_qb, f0, L_, C_, C_);
    rms_kernel<<<dim3(L_), blk, 0, stream>>>(f0, ca_nq);
    addcvt_kernel<<<g_ew, blk, 0, stream>>>(f0, nullptr, b_q, n_lc);

    gemm_kernel<0,0><<<gemm_grid(LTXT_, C_), blk, 0, stream>>>(ctxt_bf, ca_kw, ca_kb, f1, LTXT_, C_, C_);
    rms_kernel<<<dim3(LTXT_), blk, 0, stream>>>(f1, ca_nk);
    addcvt_kernel<<<dim3((LTXT_ * C_ + 255) / 256), blk, 0, stream>>>(f1, nullptr, ck_bf, LTXT_ * C_);
    gemm_kernel<0,0><<<gemm_grid(LTXT_, C_), blk, 0, stream>>>(ctxt_bf, ca_vw, ca_vb, f1, LTXT_, C_, C_);
    addcvt_kernel<<<dim3((LTXT_ * C_ + 255) / 256), blk, 0, stream>>>(f1, nullptr, cv_bf, LTXT_ * C_);

    gemm_kernel<0,0><<<gemm_grid(LIMG_, C_), blk, 0, stream>>>(cimg_bf, ca_kiw, ca_kib, f1, LIMG_, C_, C_);
    rms_kernel<<<dim3(LIMG_), blk, 0, stream>>>(f1, ca_nki);
    addcvt_kernel<<<dim3((LIMG_ * C_ + 255) / 256), blk, 0, stream>>>(f1, nullptr, cik_bf, LIMG_ * C_);
    gemm_kernel<0,0><<<gemm_grid(LIMG_, C_), blk, 0, stream>>>(cimg_bf, ca_viw, ca_vib, f1, LIMG_, C_, C_);
    addcvt_kernel<<<dim3((LIMG_ * C_ + 255) / 256), blk, 0, stream>>>(f1, nullptr, civ_bf, LIMG_ * C_);

    attn_kernel<<<dim3(L_ / 128, NH_), blk, 0, stream>>>(b_q, ck_bf, cv_bf, f1, LTXT_, scale);
    attn_kernel<<<dim3(L_ / 128, NH_), blk, 0, stream>>>(b_q, cik_bf, civ_bf, f2, LIMG_, scale);
    addcvt_kernel<<<g_ew, blk, 0, stream>>>(f1, f2, b_ao, n_lc);
    gemm_kernel<0,0><<<gemm_grid(L_, C_), blk, 0, stream>>>(b_ao, ca_ow, ca_ob, f0, L_, C_, C_);
    residual_kernel<<<g_ew, blk, 0, stream>>>(f3, f0, nullptr, f3, n_lc);

    // ---- FFN ----
    ln_mod_kernel<<<dim3(L_), blk, 0, stream>>>(f3, nullptr, nullptr,
                                                es + 4 * C_, es + 3 * C_, b_h);
    gemm_kernel<1,1><<<gemm_grid(L_, FFN_), blk, 0, stream>>>(b_h, f_w1, f_b1, b_ffn, L_, FFN_, C_);
    gemm_kernel<0,0><<<gemm_grid(L_, C_), blk, 0, stream>>>(b_ffn, f_w2, f_b2, f0, L_, C_, FFN_);
    residual_kernel<<<g_ew, blk, 0, stream>>>(f3, f0, es + 5 * C_, (float*)d_out, n_lc);
}
